// GraphAttentionLayer_33414845563580
// MI455X (gfx1250) — compile-verified, matrix operand order checked by reading, not run on hardware
//
#include <hip/hip_runtime.h>
#include <hip/hip_bf16.h>
#include <stdint.h>

// ---------------------------------------------------------------------------
// Graph attention layer for gfx1250 (MI455X), bf16 WMMA path.
//   N=4096, IN=OUT=256, H=4, HEAD=64, EDGE_DIM=64, E=131072
// Waves hold 2x2 WMMA accumulator tiles (32x32) -> 2 b128 loads per WMMA.
// ---------------------------------------------------------------------------

#define NN        4096
#define DIM       256
#define NHEAD     4
#define HEADD     64
#define EDGED     64
#define NEDGE     131072
#define LN_EPS    1e-5f

typedef __attribute__((ext_vector_type(16))) __bf16     v16bf;
typedef __attribute__((ext_vector_type(8)))  float      v8f;
typedef __attribute__((ext_vector_type(4)))  unsigned   uint4v;

// ---- helpers ---------------------------------------------------------------

__device__ __forceinline__ unsigned short f2bf(float f) {
    unsigned u = __float_as_uint(f);
    u = (u + 0x7fffu + ((u >> 16) & 1u)) >> 16;   // round-to-nearest-even
    return (unsigned short)u;
}

// Load a 16x32 bf16 WMMA A/B^T fragment from a row-major tile.
// base points at [row0][k0]; row_stride in bf16 elements; 16-byte aligned.
__device__ __forceinline__ v16bf load_frag(const unsigned short* __restrict__ base,
                                           int row_stride, int lane) {
    int r  = lane & 15;
    int kb = (lane >> 4) * 8;           // lanes 16..31 carry K+8 / K+24 halves
    const unsigned short* p = base + (size_t)r * row_stride + kb;
    union { uint4v q[2]; v16bf v; } u;
    u.q[0] = *(const uint4v*)(p);       // K = kb .. kb+7
    u.q[1] = *(const uint4v*)(p + 16);  // K = kb+16 .. kb+23
    return u.v;
}

__device__ __forceinline__ v8f wmma_bf16(v16bf a, v16bf b, v8f c) {
    return __builtin_amdgcn_wmma_f32_16x16x32_bf16(false, a, false, b,
                                                   (short)0, c, false, false);
}

// ---- stage 0: conversions --------------------------------------------------

__global__ void k_cvt(const float* __restrict__ src, unsigned short* __restrict__ dst, int n) {
    int i = blockIdx.x * blockDim.x + threadIdx.x;
    if (i < n) dst[i] = f2bf(src[i]);
}

// src f32 [R][C] -> dst bf16 [C][R]
__global__ void k_cvt_t(const float* __restrict__ src, unsigned short* __restrict__ dst,
                        int R, int C) {
    int i = blockIdx.x * blockDim.x + threadIdx.x;
    if (i < R * C) {
        int r = i / C, c = i % C;
        dst[(size_t)c * R + r] = f2bf(src[i]);
    }
}

// bf16 [R][C] -> bf16 [C][R]  (V -> Vt[h*64+d][n])
__global__ void k_t16(const unsigned short* __restrict__ src, unsigned short* __restrict__ dst,
                      int R, int C) {
    int i = blockIdx.x * blockDim.x + threadIdx.x;
    if (i < R * C) {
        int r = i / C, c = i % C;
        dst[(size_t)c * R + r] = src[i];
    }
}

// ---- stage 1: generic bf16 GEMM  C[4096][256] = A[4096][256] x Wt^T + b -----
// Block: 128 rows x 64 cols, 8 waves; wave: 32x32 (4 accumulators).

__global__ void k_gemm(const unsigned short* __restrict__ A,   // [4096][256] bf16
                       const unsigned short* __restrict__ Wt,  // [256][256] bf16, out-major
                       const float* __restrict__ bias,         // [256]
                       unsigned short* __restrict__ outB,      // bf16 out (or null)
                       float* __restrict__ outF) {             // f32 out (or null)
    int wave = threadIdx.x >> 5, lane = threadIdx.x & 31;
    int rg = wave >> 1, cg = wave & 1;
    int n0 = blockIdx.x * 128 + rg * 32;
    int o0 = blockIdx.y * 64 + cg * 32;

    const unsigned short* A0 = A  + (size_t)n0 * DIM;
    const unsigned short* A1 = A0 + (size_t)16 * DIM;
    const unsigned short* B0 = Wt + (size_t)o0 * DIM;
    const unsigned short* B1 = B0 + (size_t)16 * DIM;

    v8f c00 = {}, c01 = {}, c10 = {}, c11 = {};
#pragma unroll 2
    for (int k = 0; k < DIM; k += 32) {
        v16bf a0 = load_frag(A0 + k, DIM, lane);
        v16bf a1 = load_frag(A1 + k, DIM, lane);
        v16bf b0 = load_frag(B0 + k, DIM, lane);
        v16bf b1 = load_frag(B1 + k, DIM, lane);
        c00 = wmma_bf16(a0, b0, c00);
        c01 = wmma_bf16(a0, b1, c01);
        c10 = wmma_bf16(a1, b0, c10);
        c11 = wmma_bf16(a1, b1, c11);
    }
    int colL = lane & 15, rAdd = (lane >> 4) * 8;
    int ca = o0 + colL, cb = ca + 16;
    float ba = bias[ca], bb = bias[cb];
#pragma unroll
    for (int r = 0; r < 8; r++) {
        int rowA = n0 + r + rAdd;
        int rowB = rowA + 16;
        if (outB) {
            outB[(size_t)rowA * DIM + ca] = f2bf(c00[r] + ba);
            outB[(size_t)rowA * DIM + cb] = f2bf(c01[r] + bb);
            outB[(size_t)rowB * DIM + ca] = f2bf(c10[r] + ba);
            outB[(size_t)rowB * DIM + cb] = f2bf(c11[r] + bb);
        } else {
            outF[(size_t)rowA * DIM + ca] = c00[r] + ba;
            outF[(size_t)rowA * DIM + cb] = c01[r] + bb;
            outF[(size_t)rowB * DIM + ca] = c10[r] + ba;
            outF[(size_t)rowB * DIM + cb] = c11[r] + bb;
        }
    }
}

// ---- stage 2: scores S[h][n][m] = 0.125*<Q[n,h,:],K[m,h,:]> + 10*adj[n][m] --
// Block: 128 (n) x 64 (m) per head; wave: 32x32.

__global__ void k_scores(const unsigned short* __restrict__ Qb,
                         const unsigned short* __restrict__ Kb,
                         const float* __restrict__ adj,
                         float* __restrict__ S) {
    int wave = threadIdx.x >> 5, lane = threadIdx.x & 31;
    int rg = wave >> 1, cg = wave & 1;
    int h  = blockIdx.z;
    int n0 = blockIdx.x * 128 + rg * 32;
    int m0 = blockIdx.y * 64 + cg * 32;

    const unsigned short* A0 = Qb + (size_t)n0 * DIM + h * HEADD;
    const unsigned short* A1 = A0 + (size_t)16 * DIM;
    const unsigned short* B0 = Kb + (size_t)m0 * DIM + h * HEADD;
    const unsigned short* B1 = B0 + (size_t)16 * DIM;

    v8f c00 = {}, c01 = {}, c10 = {}, c11 = {};
#pragma unroll
    for (int k = 0; k < HEADD; k += 32) {
        v16bf a0 = load_frag(A0 + k, DIM, lane);
        v16bf a1 = load_frag(A1 + k, DIM, lane);
        v16bf b0 = load_frag(B0 + k, DIM, lane);
        v16bf b1 = load_frag(B1 + k, DIM, lane);
        c00 = wmma_bf16(a0, b0, c00);
        c01 = wmma_bf16(a0, b1, c01);
        c10 = wmma_bf16(a1, b0, c10);
        c11 = wmma_bf16(a1, b1, c11);
    }
    int colL = lane & 15, rAdd = (lane >> 4) * 8;
    int ma = m0 + colL, mb = ma + 16;
#pragma unroll
    for (int r = 0; r < 8; r++) {
        int rowA = n0 + r + rAdd;
        int rowB = rowA + 16;
        size_t sA = ((size_t)h * NN + rowA) * NN;
        size_t sB = ((size_t)h * NN + rowB) * NN;
        size_t aA = (size_t)rowA * NN;
        size_t aB = (size_t)rowB * NN;
        S[sA + ma] = c00[r] * 0.125f + 10.0f * adj[aA + ma];
        S[sA + mb] = c01[r] * 0.125f + 10.0f * adj[aA + mb];
        S[sB + ma] = c10[r] * 0.125f + 10.0f * adj[aB + ma];
        S[sB + mb] = c11[r] * 0.125f + 10.0f * adj[aB + mb];
    }
}

// ---- stage 3: per-edge additive bias, scatter-add (duplicates -> atomics) ---

__global__ void k_edge(const float* __restrict__ ef, const int* __restrict__ ei,
                       const float* __restrict__ We, const float* __restrict__ be,
                       float* __restrict__ S) {
    int e = blockIdx.x * blockDim.x + threadIdx.x;
    if (e >= NEDGE) return;
    float acc0 = be[0], acc1 = be[1], acc2 = be[2], acc3 = be[3];
    const float* row = ef + (size_t)e * EDGED;
#pragma unroll 8
    for (int i = 0; i < EDGED; i++) {
        float x = row[i];
        acc0 += x * We[i * 4 + 0];
        acc1 += x * We[i * 4 + 1];
        acc2 += x * We[i * 4 + 2];
        acc3 += x * We[i * 4 + 3];
    }
    int src = ei[e * 2 + 0], tgt = ei[e * 2 + 1];
    size_t base = (size_t)src * NN + tgt;
    atomicAdd(&S[0 * (size_t)NN * NN + base], acc0);
    atomicAdd(&S[1 * (size_t)NN * NN + base], acc1);
    atomicAdd(&S[2 * (size_t)NN * NN + base], acc2);
    atomicAdd(&S[3 * (size_t)NN * NN + base], acc3);
}

// ---- stage 4: row softmax over m, write bf16 probs IN PLACE over f32 row ----

__global__ void k_softmax(float* __restrict__ S) {
    int rowid = blockIdx.x;                  // h*NN + n
    float* srow = S + (size_t)rowid * NN;
    unsigned short* prow = (unsigned short*)srow;
    int t = threadIdx.x;                     // 256 threads, 16 elems each

    float v[16];
    float mx = -3.4e38f;
#pragma unroll
    for (int k = 0; k < 16; k++) { v[k] = srow[t + k * 256]; mx = fmaxf(mx, v[k]); }

    __shared__ float red[256];
    red[t] = mx; __syncthreads();
    for (int s = 128; s > 0; s >>= 1) { if (t < s) red[t] = fmaxf(red[t], red[t + s]); __syncthreads(); }
    mx = red[0]; __syncthreads();

    float sum = 0.0f;
#pragma unroll
    for (int k = 0; k < 16; k++) { v[k] = __expf(v[k] - mx); sum += v[k]; }
    red[t] = sum; __syncthreads();
    for (int s = 128; s > 0; s >>= 1) { if (t < s) red[t] += red[t + s]; __syncthreads(); }
    float inv = 1.0f / red[0];
    __syncthreads();                         // all reads of srow complete before overwrite
#pragma unroll
    for (int k = 0; k < 16; k++) prow[t + k * 256] = f2bf(v[k] * inv);
}

// ---- stage 5: O[n][h*64+d] = sum_m P[h][n][m] * Vt[h][d][m] -----------------
// Block: 128 (n) x 64 (d, full head) per head; wave: 32x32.

__global__ void k_av(const unsigned short* __restrict__ Pb,  // bf16 @ S row bases, stride 8192
                     const unsigned short* __restrict__ Vt,  // [h*64+d][NN] bf16
                     unsigned short* __restrict__ Ob) {      // [NN][256] bf16
    int wave = threadIdx.x >> 5, lane = threadIdx.x & 31;
    int rg = wave >> 1, cg = wave & 1;
    int h  = blockIdx.y;
    int n0 = blockIdx.x * 128 + rg * 32;
    int d0 = cg * 32;

    const unsigned short* A0 = Pb + ((size_t)h * NN + n0) * (2 * NN);
    const unsigned short* A1 = A0 + (size_t)16 * (2 * NN);
    const unsigned short* B0 = Vt + ((size_t)h * HEADD + d0) * NN;
    const unsigned short* B1 = B0 + (size_t)16 * NN;

    v8f c00 = {}, c01 = {}, c10 = {}, c11 = {};
#pragma unroll 2
    for (int m = 0; m < NN; m += 32) {
        v16bf a0 = load_frag(A0 + m, 2 * NN, lane);
        v16bf a1 = load_frag(A1 + m, 2 * NN, lane);
        v16bf b0 = load_frag(B0 + m, NN, lane);
        v16bf b1 = load_frag(B1 + m, NN, lane);
        c00 = wmma_bf16(a0, b0, c00);
        c01 = wmma_bf16(a0, b1, c01);
        c10 = wmma_bf16(a1, b0, c10);
        c11 = wmma_bf16(a1, b1, c11);
    }
    int colL = lane & 15, rAdd = (lane >> 4) * 8;
    int da = h * HEADD + d0 + colL, db = da + 16;
#pragma unroll
    for (int r = 0; r < 8; r++) {
        int rowA = n0 + r + rAdd;
        int rowB = rowA + 16;
        Ob[(size_t)rowA * DIM + da] = f2bf(c00[r]);
        Ob[(size_t)rowA * DIM + db] = f2bf(c01[r]);
        Ob[(size_t)rowB * DIM + da] = f2bf(c10[r]);
        Ob[(size_t)rowB * DIM + db] = f2bf(c11[r]);
    }
}

// ---- stage 6: residual + LayerNorm -----------------------------------------

__global__ void k_ln(const float* __restrict__ Y, const float* __restrict__ X,
                     const float* __restrict__ gamma, const float* __restrict__ beta,
                     float* __restrict__ out) {
    int n = blockIdx.x, t = threadIdx.x;     // 256 threads = 256 columns
    float x = Y[(size_t)n * DIM + t] + X[(size_t)n * DIM + t];
    __shared__ float red[256];
    red[t] = x; __syncthreads();
    for (int s = 128; s > 0; s >>= 1) { if (t < s) red[t] += red[t + s]; __syncthreads(); }
    float mu = red[0] * (1.0f / DIM);
    __syncthreads();
    float d = x - mu;
    red[t] = d * d; __syncthreads();
    for (int s = 128; s > 0; s >>= 1) { if (t < s) red[t] += red[t + s]; __syncthreads(); }
    float var = red[0] * (1.0f / DIM);
    out[(size_t)n * DIM + t] = gamma[t] * d * rsqrtf(var + LN_EPS) + beta[t];
}

// ---------------------------------------------------------------------------

extern "C" void kernel_launch(void* const* d_in, const int* in_sizes, int n_in,
                              void* d_out, int out_size, void* d_ws, size_t ws_size,
                              hipStream_t stream) {
    (void)in_sizes; (void)n_in; (void)out_size; (void)ws_size;

    const float* X    = (const float*)d_in[0];
    const float* adj  = (const float*)d_in[1];
    const float* ef   = (const float*)d_in[2];
    const int*   ei   = (const int*)  d_in[3];
    const float* Wq   = (const float*)d_in[4];
    const float* bq   = (const float*)d_in[5];
    const float* Wk   = (const float*)d_in[6];
    const float* bk   = (const float*)d_in[7];
    const float* Wv   = (const float*)d_in[8];
    const float* bv   = (const float*)d_in[9];
    const float* We   = (const float*)d_in[10];
    const float* be   = (const float*)d_in[11];
    const float* Wo   = (const float*)d_in[12];
    const float* bo   = (const float*)d_in[13];
    const float* lng  = (const float*)d_in[14];
    const float* lnb  = (const float*)d_in[15];
    float*       out  = (float*)d_out;

    // workspace carve-up
    uint8_t* w = (uint8_t*)d_ws;
    const size_t S_BYTES  = (size_t)NHEAD * NN * NN * sizeof(float);   // 256 MB
    const size_t BF_NODES = (size_t)NN * DIM * sizeof(unsigned short); // 2 MB
    const size_t BF_W     = (size_t)DIM * DIM * sizeof(unsigned short);

    float*          S   = (float*)w;                    w += S_BYTES;
    unsigned short* Xb  = (unsigned short*)w;           w += BF_NODES;
    unsigned short* Qb  = (unsigned short*)w;           w += BF_NODES;
    unsigned short* Kb  = (unsigned short*)w;           w += BF_NODES;
    unsigned short* Vb  = (unsigned short*)w;           w += BF_NODES;
    unsigned short* Vt  = (unsigned short*)w;           w += BF_NODES;
    unsigned short* Ob  = (unsigned short*)w;           w += BF_NODES;
    unsigned short* Wqt = (unsigned short*)w;           w += BF_W;
    unsigned short* Wkt = (unsigned short*)w;           w += BF_W;
    unsigned short* Wvt = (unsigned short*)w;           w += BF_W;
    unsigned short* Wot = (unsigned short*)w;           w += BF_W;
    float*          Y   = (float*)w;                    w += (size_t)NN * DIM * sizeof(float);

    // 0) conversions
    k_cvt  <<<dim3((NN * DIM + 255) / 256), 256, 0, stream>>>(X, Xb, NN * DIM);
    k_cvt_t<<<dim3((DIM * DIM + 255) / 256), 256, 0, stream>>>(Wq, Wqt, DIM, DIM);
    k_cvt_t<<<dim3((DIM * DIM + 255) / 256), 256, 0, stream>>>(Wk, Wkt, DIM, DIM);
    k_cvt_t<<<dim3((DIM * DIM + 255) / 256), 256, 0, stream>>>(Wv, Wvt, DIM, DIM);
    k_cvt_t<<<dim3((DIM * DIM + 255) / 256), 256, 0, stream>>>(Wo, Wot, DIM, DIM);

    // 1) Q/K/V projections (bf16 WMMA)
    k_gemm<<<dim3(NN / 128, DIM / 64), 256, 0, stream>>>(Xb, Wqt, bq, Qb, nullptr);
    k_gemm<<<dim3(NN / 128, DIM / 64), 256, 0, stream>>>(Xb, Wkt, bk, Kb, nullptr);
    k_gemm<<<dim3(NN / 128, DIM / 64), 256, 0, stream>>>(Xb, Wvt, bv, Vb, nullptr);
    k_t16 <<<dim3((NN * DIM + 255) / 256), 256, 0, stream>>>(Vb, Vt, NN, DIM);

    // 2) scores = scale*QK^T + 10*adj
    k_scores<<<dim3(NN / 128, NN / 64, NHEAD), 256, 0, stream>>>(Qb, Kb, adj, S);

    // 3) edge bias scatter-add
    k_edge<<<dim3(NEDGE / 256), 256, 0, stream>>>(ef, ei, We, be, S);

    // 4) softmax over keys, bf16 probs written in place
    k_softmax<<<dim3(NHEAD * NN), 256, 0, stream>>>(S);

    // 5) O = P @ V (bf16 WMMA over m=4096)
    k_av<<<dim3(NN / 128, NHEAD), 256, 0, stream>>>((const unsigned short*)S, Vt, Ob);

    // 6) output projection, then residual + LayerNorm
    k_gemm<<<dim3(NN / 128, DIM / 64), 256, 0, stream>>>(Ob, Wot, bo, nullptr, Y);
    k_ln  <<<dim3(NN), 256, 0, stream>>>(Y, X, lng, lnb, out);
}
